// GeodesicConv_50019189129841
// MI455X (gfx1250) — matrix operand order repaired and checked by reading, not compiled, not run on hardware
//
#include <hip/hip_runtime.h>
#include <hip/hip_bf16.h>

// ---- CDNA5 WMMA types -------------------------------------------------------
typedef __attribute__((ext_vector_type(16))) __bf16 v16bf;
typedef __attribute__((ext_vector_type(8)))  float  v8f;

struct __align__(16) U4 { unsigned x, y, z, w; };
struct __align__(32) U8 { U4 lo, hi; };

__device__ __forceinline__ unsigned pk_bf16(float a, float b) {
  // round-to-nearest-even f32 -> bf16, packed pair (a in [15:0])
  unsigned ua = __builtin_bit_cast(unsigned, a);
  unsigned ub = __builtin_bit_cast(unsigned, b);
  ua += 0x7FFFu + ((ua >> 16) & 1u);
  ub += 0x7FFFu + ((ub >> 16) & 1u);
  return (ua >> 16) | (ub & 0xFFFF0000u);
}

#define NRINGS  8
#define NDIRS   8
#define NCH     16
#define NF      16
#define WAVES   4
#define VPW     2                      // vertices per wave (16x16 tile = 2v x 8d)
#define VPB     (WAVES * VPW)          // vertices per block
#define ZSLOTS  9                      // 8 rings + 1 "center" slot holding raw y
#define ZSTRIDE (ZSLOTS * NDIRS * NCH) // ushorts per vertex tile (1152)
#define NCHUNK  33                     // 1024 conv K + 32 (center fused, half zero)

__global__ __launch_bounds__(WAVES * 32)
void geodesic_conv_wmma(const float* __restrict__ y,
                        const int*   __restrict__ contrib,
                        const float* __restrict__ wbary,
                        const int*   __restrict__ ang,
                        const float* __restrict__ kern,     // (8,8,16,16) == [K=1024][F=16]
                        const float* __restrict__ ckern,    // (16,16)
                        const float* __restrict__ bias,     // (16,)
                        float*       __restrict__ out,      // (B, NV, 16)
                        int NV)
{
  __shared__ alignas(16) unsigned short zbuf[WAVES * VPW * ZSTRIDE];   // 18432 B (bf16 z tiles)
  __shared__ alignas(16) unsigned int   bfrag[NCHUNK * 32 * 8];        // 33792 B (B fragments)

  const int tid  = threadIdx.x;
  const int wave = tid >> 5;
  const int lane = tid & 31;
  const int b    = blockIdx.y;
  const int vblk = blockIdx.x * VPB;

  // ---- prefetch this block's gather metadata (global_prefetch_b8) ----------
  {
    const long mbase = (long)vblk * (NRINGS * NDIRS * 3) * 4;          // byte offset
    const int  nbytes = VPB * NRINGS * NDIRS * 3 * 4;                  // 6144 B per array
    int off = tid * 128;
    if (off < nbytes) {
      __builtin_prefetch((const char*)contrib + mbase + off, 0, 3);
      __builtin_prefetch((const char*)wbary   + mbase + off, 0, 3);
      __builtin_prefetch((const char*)ang     + mbase + off, 0, 3);
    }
  }

  // ---- build bf16 B fragments in exact v_wmma lane layout (once per block) -
  // B matrix rows: K<1024 -> conv kernel (layout is literally kern[K*16+N]);
  //                1024<=K<1040 -> center_kernel; K>=1040 -> zero padding.
  for (int e = tid; e < NCHUNK * 32; e += WAVES * 32) {
    const int chunk = e >> 5;
    const int ln    = e & 31;
    const int N     = ln & 15;
    const int kb    = chunk * 32 + ((ln >> 4) << 3);  // lanes 16-31 hold K+8..K+15, K+24..K+31
    unsigned vals[8];
#pragma unroll
    for (int p = 0; p < 8; ++p) {
      const int K0 = kb + 2 * p + ((p >= 4) ? 8 : 0), K1 = K0 + 1;
      const float f0 = (K0 < 1024) ? kern[K0 * NF + N]
                     : ((K0 < 1040) ? ckern[(K0 - 1024) * NF + N] : 0.0f);
      const float f1 = (K1 < 1024) ? kern[K1 * NF + N]
                     : ((K1 < 1040) ? ckern[(K1 - 1024) * NF + N] : 0.0f);
      vals[p] = pk_bf16(f0, f1);
    }
    unsigned* dst = &bfrag[e * 8];
    *(U4*)(dst)     = U4{vals[0], vals[1], vals[2], vals[3]};
    *(U4*)(dst + 4) = U4{vals[4], vals[5], vals[6], vals[7]};
  }

  // ---- stage 1: fp32 gather + barycentric blend -> bf16 z tiles in LDS -----
  {
    const long ybatch = (long)b * NV * NDIRS * NCH;
    for (int it = 0; it < 5; ++it) {
      const int row = it * 32 + lane;                 // 128 z rows + 16 y rows per wave
      if (row < 128) {
        const int vsub = row >> 6;
        const int rem  = row & 63;
        const int r = rem >> 3, d = rem & 7;
        int vg = vblk + wave * VPW + vsub;
        if (vg >= NV) vg = NV - 1;
        const long mb = ((long)vg * 64 + r * 8 + d) * 3;
        const int   c0 = contrib[mb + 0], c1 = contrib[mb + 1], c2 = contrib[mb + 2];
        const int   a0 = ang[mb + 0],     a1 = ang[mb + 1],     a2 = ang[mb + 2];
        const float w0 = wbary[mb + 0],   w1 = wbary[mb + 1],   w2 = wbary[mb + 2];
        const float4* p0 = (const float4*)(y + ybatch + (long)(c0 * NDIRS + a0) * NCH);
        const float4* p1 = (const float4*)(y + ybatch + (long)(c1 * NDIRS + a1) * NCH);
        const float4* p2 = (const float4*)(y + ybatch + (long)(c2 * NDIRS + a2) * NCH);
        unsigned* zp = (unsigned*)&zbuf[(wave * VPW + vsub) * ZSTRIDE + (r * NDIRS + d) * NCH];
#pragma unroll
        for (int q = 0; q < 4; ++q) {
          const float4 g0 = p0[q], g1 = p1[q], g2 = p2[q];
          const float sx = w0 * g0.x + w1 * g1.x + w2 * g2.x;
          const float sy = w0 * g0.y + w1 * g1.y + w2 * g2.y;
          const float sz = w0 * g0.z + w1 * g1.z + w2 * g2.z;
          const float sw = w0 * g0.w + w1 * g1.w + w2 * g2.w;
          zp[q * 2 + 0] = pk_bf16(sx, sy);
          zp[q * 2 + 1] = pk_bf16(sz, sw);
        }
      } else if (row < 144) {                         // raw y into slot r=8 (center term)
        const int rem  = row - 128;
        const int vsub = rem >> 3, d = rem & 7;
        int vg = vblk + wave * VPW + vsub;
        if (vg >= NV) vg = NV - 1;
        const float4* p = (const float4*)(y + ybatch + ((long)vg * NDIRS + d) * NCH);
        unsigned* zp = (unsigned*)&zbuf[(wave * VPW + vsub) * ZSTRIDE + (8 * NDIRS + d) * NCH];
#pragma unroll
        for (int q = 0; q < 4; ++q) {
          const float4 g = p[q];
          zp[q * 2 + 0] = pk_bf16(g.x, g.y);
          zp[q * 2 + 1] = pk_bf16(g.z, g.w);
        }
      }
    }
  }

  __syncthreads();

  // ---- stage 2: 33 chained v_wmma_f32_16x16x32_bf16 ------------------------
  // A row M = vsub*8 + d_out; K = r*128 + d'*16 + c ; A[M][K] = z[vsub][r][(d_out+d')&7][c]
  v8f acc = {};
  const int m     = lane & 15;        // A-matrix row owned by this lane
  const int vsub  = m >> 3;
  const int dout  = m & 7;
  const int khalf = lane >> 4;        // lanes 16-31 hold K offsets +8 within each 16-group
  const unsigned short* zB = &zbuf[(wave * VPW + vsub) * ZSTRIDE];

#pragma unroll
  for (int kk = 0; kk < NCHUNK; ++kk) {
    const int k0  = kk * 32 + khalf * 8;
    const int g0  = k0 >> 4;
    const int c0  = k0 & 15;
    const int g1  = g0 + 1;
    const int r0  = g0 >> 3, dp0 = g0 & 7;
    const int r1  = g1 >> 3, dp1 = g1 & 7;
    const int de0 = (dout + dp0) & 7;
    const int de1 = (dout + dp1) & 7;
    U8 a8, b8;
    a8.lo = *(const U4*)(zB + (r0 * NDIRS + de0) * NCH + c0);   // VGPRs 0-3 (K=k0..k0+7)
    a8.hi = *(const U4*)(zB + (r1 * NDIRS + de1) * NCH + c0);   // VGPRs 4-7 (K=k0+16..)
    const unsigned* bp = &bfrag[(kk * 32 + lane) * 8];
    b8.lo = *(const U4*)(bp);
    b8.hi = *(const U4*)(bp + 4);
    const v16bf av = __builtin_bit_cast(v16bf, a8);
    const v16bf bv = __builtin_bit_cast(v16bf, b8);
    acc = __builtin_amdgcn_wmma_f32_16x16x32_bf16(false, av, false, bv,
                                                  (short)0, acc, false, false);
  }

  // ---- epilogue: relu(acc + bias), max over 8 dirs, one store per lane -----
  // VGPR i holds M = (lane<16 ? i : 8+i)  => lanes 0-15: vertex 0, lanes 16-31: vertex 1
  const float bi = bias[m];
  float best = 0.0f;                  // relu floor
#pragma unroll
  for (int i = 0; i < 8; ++i) best = fmaxf(best, acc[i] + bi);

  const int vg = vblk + wave * VPW + (lane >> 4);
  if (vg < NV)
    out[((long)b * NV + vg) * NF + m] = best;
}

// ---- host launcher ----------------------------------------------------------
extern "C" void kernel_launch(void* const* d_in, const int* in_sizes, int n_in,
                              void* d_out, int out_size, void* d_ws, size_t ws_size,
                              hipStream_t stream) {
  const float* y       = (const float*)d_in[0];
  const int*   contrib = (const int*)  d_in[1];
  const float* wbary   = (const float*)d_in[2];
  const int*   ang     = (const int*)  d_in[3];
  const float* kern    = (const float*)d_in[4];
  const float* ckern   = (const float*)d_in[5];
  const float* bias    = (const float*)d_in[6];
  float*       out     = (float*)d_out;

  const int NV = in_sizes[1] / (NRINGS * NDIRS * 3);      // contributors element count
  const int B  = in_sizes[0] / (NV * NDIRS * NCH);        // y element count

  dim3 grid((NV + VPB - 1) / VPB, B);
  geodesic_conv_wmma<<<grid, WAVES * 32, 0, stream>>>(
      y, contrib, wbary, ang, kern, ckern, bias, out, NV);
}
